// DyVM_83897891160362
// MI455X (gfx1250) — matrix operand order repaired and checked by reading, not compiled, not run on hardware
//
#include <hip/hip_runtime.h>
#include <hip/hip_bf16.h>
#include <math.h>

// ---------------------------------------------------------------------------
// DyVM forward for MI455X (gfx1250). All dense linear algebra goes through
// v_wmma_f32_16x16x32_f16. GEMM operands are staged as zero-padded f16 so the
// WMMA inner loop is pure b128 loads + wmma (no predication, no waits-per-elem).
// ---------------------------------------------------------------------------

typedef __attribute__((ext_vector_type(16))) _Float16 v16h;
typedef __attribute__((ext_vector_type(8)))  _Float16 v8h;
typedef __attribute__((ext_vector_type(8)))  float    v8f;

#define EPI_NONE     0
#define EPI_BIAS     1
#define EPI_GELU     2
#define EPI_SOFTPLUS 3

// ---------------- f32 -> f16 staging with zero padding --------------------
// dst[r, c] (r < rowsP implicit via total, c < colsP) = (r<rows && c<cols) ? src[r*ld+c] : 0
__global__ void cvt_pad_f16(const float* __restrict__ src, int ld, int rows, int cols,
                            _Float16* __restrict__ dst, int colsP, int total)
{
    int idx = blockIdx.x * blockDim.x + threadIdx.x;
    if (idx >= total) return;
    int c = idx % colsP;
    int r = idx / colsP;
    float v = (r < rows && c < cols) ? src[(size_t)r * ld + c] : 0.f;
    dst[idx] = (_Float16)v;
}

// ---------------- WMMA GEMM on padded f16:  C = epi( A * W^T + bias ) ------
// A: [M, Kp] f16 row-major (M multiple of 16, Kp multiple of 32)
// W: [Np, Kp] f16 row-major (Np multiple of 16, rows >= N zero-padded)
// one wave == one 16x16 tile; inner loop is branch-free.
__global__ void gemm_wmma_f16(const _Float16* __restrict__ A,
                              const _Float16* __restrict__ W,
                              const float* __restrict__ bias,
                              float* __restrict__ C, int ldc,
                              int M, int N, int Kp, int epi)
{
    const int lane  = threadIdx.x;                       // 0..31
    const int mtile = blockIdx.x * blockDim.y + threadIdx.y;
    const int m0    = mtile << 4;
    if (m0 >= M) return;                                  // wave-uniform
    const int n0   = blockIdx.y << 4;
    const int half = lane >> 4;                           // 0 | 1
    const int l15  = lane & 15;

    // A-fragment (ISA 16-bit A 16x32): lanes<16 hold K {0..7,16..23}, lanes>=16
    // hold K {8..15,24..31} of each 32-block; row = lane&15.
    const _Float16* __restrict__ Arow = A + (size_t)(m0 + l15) * Kp + (half << 3);
    // B-fragment (32x16): lane = column, same K order as A: {0..7,16..23} / {8..15,24..31}.
    const _Float16* __restrict__ Wrow = W + (size_t)(n0 + l15) * Kp + (half << 3);

    v8f acc = {0.f, 0.f, 0.f, 0.f, 0.f, 0.f, 0.f, 0.f};

    for (int kb = 0; kb < Kp; kb += 32) {
        v8h a0 = *(const v8h*)(Arow + kb);        // 16B aligned
        v8h a1 = *(const v8h*)(Arow + kb + 16);
        v8h b0 = *(const v8h*)(Wrow + kb);
        v8h b1 = *(const v8h*)(Wrow + kb + 16);
        v16h a = __builtin_shufflevector(a0, a1, 0,1,2,3,4,5,6,7,8,9,10,11,12,13,14,15);
        v16h b = __builtin_shufflevector(b0, b1, 0,1,2,3,4,5,6,7,8,9,10,11,12,13,14,15);
        acc = __builtin_amdgcn_wmma_f32_16x16x32_f16(
            /*neg_a=*/false, a, /*neg_b=*/false, b,
            /*c_mod=*/(short)0, acc, /*reuse_a=*/false, /*reuse_b=*/false);
    }

    const int ncol = n0 + l15;
    if (ncol >= N) return;
    float bv = (epi != EPI_NONE && bias) ? bias[ncol] : 0.f;
    #pragma unroll
    for (int r = 0; r < 8; ++r) {
        float v = acc[r] + bv;
        if (epi == EPI_GELU) {
            v = 0.5f * v * (1.f + erff(v * 0.70710678118654752f));
        } else if (epi == EPI_SOFTPLUS) {
            v = (v > 20.f) ? v : log1pf(expf(v));
        }
        int mm = m0 + (half << 3) + r;   // D layout: VGPR r -> M = r + half*8
        C[(size_t)mm * (size_t)ldc + ncol] = v;
    }
}

// ---------------- im2col for the 16x16/stride16 patch embedding (emit f16)
__global__ void im2col_patch_f16(const float* __restrict__ x, _Float16* __restrict__ out, int total)
{
    int idx = blockIdx.x * blockDim.x + threadIdx.x;
    if (idx >= total) return;
    int k = idx % 768;
    int m = idx / 768;          // b*196 + t
    int t = m % 196, b = m / 196;
    int c  = k >> 8;
    int py = (k >> 4) & 15;
    int px = k & 15;
    int ty = t / 14, tx = t % 14;
    out[idx] = (_Float16)x[(((size_t)b * 3 + c) * 224 + (ty * 16 + py)) * 224 + (tx * 16 + px)];
}

// ---------------- hidden = cat(cls, tok) + pos_embed
__global__ void assemble_hidden(const float* __restrict__ tok, const float* __restrict__ cls,
                                const float* __restrict__ pos, float* __restrict__ hid, int total)
{
    int idx = blockIdx.x * blockDim.x + threadIdx.x;
    if (idx >= total) return;
    int e = idx % 192;
    int n = idx / 192;
    int l = n % 197;
    float v = (l == 0) ? cls[e] : tok[((size_t)(n / 197) * 196 + (l - 1)) * 192 + e];
    hid[idx] = v + pos[l * 192 + e];
}

// ---------------- residual += hidden (or = hidden); out = rmsnorm(residual, w)
__global__ void add_rmsnorm(const float* __restrict__ hid, float* __restrict__ res,
                            const float* __restrict__ w, float* __restrict__ out,
                            int nrows, int first)
{
    int row  = blockIdx.x * (blockDim.x >> 5) + (threadIdx.x >> 5);
    int lane = threadIdx.x & 31;
    if (row >= nrows) return;
    const float* hp = hid + (size_t)row * 192;
    float*       rp = res + (size_t)row * 192;
    float vals[6]; float ss = 0.f;
    #pragma unroll
    for (int j = 0; j < 6; ++j) {
        int e = lane + j * 32;
        float v = hp[e] + (first ? 0.f : rp[e]);
        vals[j] = v; ss += v * v;
    }
    #pragma unroll
    for (int off = 16; off > 0; off >>= 1) ss += __shfl_xor(ss, off, 32);
    float sc = rsqrtf(ss / 192.f + 1e-5f);
    #pragma unroll
    for (int j = 0; j < 6; ++j) {
        int e = lane + j * 32;
        rp[e] = vals[j];
        out[(size_t)row * 192 + e] = vals[j] * sc * w[e];
    }
}

// ---------------- predictor layernorm over (hidden+residual) of non-cls tokens
__global__ void pred_layernorm(const float* __restrict__ hid, const float* __restrict__ res,
                               int L, const float* __restrict__ w, const float* __restrict__ b,
                               float* __restrict__ out, int nrows)
{
    int row  = blockIdx.x * (blockDim.x >> 5) + (threadIdx.x >> 5);
    int lane = threadIdx.x & 31;
    if (row >= nrows) return;
    int no = L - 1;
    size_t src = (size_t)(row / no) * L + 1 + (row % no);
    const float* hp = hid + src * 192;
    const float* rp = res + src * 192;
    float vals[6]; float s = 0.f;
    #pragma unroll
    for (int j = 0; j < 6; ++j) { float v = hp[lane + j*32] + rp[lane + j*32]; vals[j] = v; s += v; }
    #pragma unroll
    for (int off = 16; off > 0; off >>= 1) s += __shfl_xor(s, off, 32);
    float mu = s / 192.f, s2 = 0.f;
    #pragma unroll
    for (int j = 0; j < 6; ++j) { float d = vals[j] - mu; s2 += d * d; }
    #pragma unroll
    for (int off = 16; off > 0; off >>= 1) s2 += __shfl_xor(s2, off, 32);
    float inv = rsqrtf(s2 / 192.f + 1e-5f);
    #pragma unroll
    for (int j = 0; j < 6; ++j) {
        int e = lane + j * 32;
        out[(size_t)row * 192 + e] = (vals[j] - mu) * inv * w[e] + b[e];
    }
}

// ---------------- causal depthwise conv (k=4, left pad 3) + silu; optional seq reverse
__global__ void conv_silu(const float* __restrict__ xz, int L,
                          const float* __restrict__ cw, const float* __restrict__ cb,
                          float* __restrict__ xs, int total, int rev)
{
    int idx = blockIdx.x * blockDim.x + threadIdx.x;
    if (idx >= total) return;
    int d = idx % 384;
    int n = idx / 384;
    int l = n % L, b = n / L;
    float acc = cb[d];
    #pragma unroll
    for (int j = 0; j < 4; ++j) {
        int li = l - 3 + j;
        if (li < 0) continue;
        int lsrc = rev ? (L - 1 - li) : li;
        acc += cw[d * 4 + j] * xz[((size_t)b * L + lsrc) * 768 + d];
    }
    xs[idx] = acc / (1.f + expf(-acc));
}

// ---------------- selective scan: one thread per (b, d) channel, 16 states in regs
__global__ void ssm_scan(const float* __restrict__ xs, const float* __restrict__ dbc,
                         const float* __restrict__ dt, const float* __restrict__ A_log,
                         const float* __restrict__ Dp, float* __restrict__ ys, int L)
{
    int gid = blockIdx.x * blockDim.x + threadIdx.x;
    if (gid >= 128 * 384) return;
    int d = gid % 384, b = gid / 384;
    float Anb[16], h[16];
    #pragma unroll
    for (int s = 0; s < 16; ++s) { Anb[s] = -expf(A_log[d * 16 + s]); h[s] = 0.f; }
    float Dd = Dp[d];
    for (int l = 0; l < L; ++l) {
        size_t n = (size_t)b * L + l;
        float dtv = dt[n * 384 + d];
        float xv  = xs[n * 384 + d];
        float dx  = dtv * xv;
        const float* BC = dbc + n * 44;
        float y = 0.f;
        #pragma unroll
        for (int s = 0; s < 16; ++s) {
            h[s] = expf(dtv * Anb[s]) * h[s] + dx * BC[12 + s];
            y += h[s] * BC[28 + s];
        }
        ys[n * 384 + d] = y + xv * Dd;
    }
}

// ---------------- yy = (yf + reverse(yb)) * silu(z)
__global__ void combine_gate(const float* __restrict__ yf, const float* __restrict__ yb,
                             const float* __restrict__ xz, int L,
                             float* __restrict__ yy, int total)
{
    int idx = blockIdx.x * blockDim.x + threadIdx.x;
    if (idx >= total) return;
    int d = idx % 384;
    int n = idx / 384;
    int l = n % L, b = n / L;
    size_t nf = (size_t)b * L + l;
    size_t nr = (size_t)b * L + (L - 1 - l);
    float z = xz[nf * 768 + 384 + d];
    yy[idx] = (yf[nf * 384 + d] + yb[nr * 384 + d]) * (z / (1.f + expf(-z)));
}

// ---------------- predictor global pool over tokens of xp[:, 96:192]
__global__ void pred_pool(const float* __restrict__ xp, int no, float* __restrict__ glob, int total)
{
    int idx = blockIdx.x * blockDim.x + threadIdx.x;
    if (idx >= total) return;
    int e = idx % 96, b = idx / 96;
    float s = 0.f;
    for (int t = 0; t < no; ++t) s += xp[((size_t)b * no + t) * 192 + 96 + e];
    glob[idx] = s / (float)no;
}

__global__ void pred_feat(const float* __restrict__ xp, const float* __restrict__ glob,
                          int no, float* __restrict__ feat, int total)
{
    int idx = blockIdx.x * blockDim.x + threadIdx.x;
    if (idx >= total) return;
    int e = idx % 192;
    int n = idx / 192;
    feat[idx] = (e < 96) ? xp[(size_t)n * 192 + e] : glob[(n / no) * 96 + (e - 96)];
}

__global__ void pred_score(const float* __restrict__ logit2, float* __restrict__ score, int total)
{
    int n = blockIdx.x * blockDim.x + threadIdx.x;
    if (n >= total) return;
    float l0 = logit2[(size_t)n * 2], l1 = logit2[(size_t)n * 2 + 1];
    float mx = fmaxf(l0, l1);
    score[n] = l0 - (mx + logf(expf(l0 - mx) + expf(l1 - mx)));
}

// ---------------- top-k (per batch row) with ascending-index compaction
__global__ void topk_sorted(const float* __restrict__ score, int no, int k, int* __restrict__ idxmap)
{
    __shared__ float sc[224];
    __shared__ int   sel[224];
    __shared__ float red[256];
    __shared__ int   redi[256];
    int b = blockIdx.x, t = threadIdx.x;
    if (t < no) { sc[t] = score[(size_t)b * no + t]; sel[t] = 0; }
    __syncthreads();
    for (int iter = 0; iter < k; ++iter) {
        float v = (t < no && !sel[t]) ? sc[t] : -3.0e38f;
        red[t] = v; redi[t] = t;
        __syncthreads();
        for (int off = 128; off > 0; off >>= 1) {
            if (t < off && red[t + off] > red[t]) { red[t] = red[t + off]; redi[t] = redi[t + off]; }
            __syncthreads();
        }
        if (t == 0) sel[redi[0]] = 1;
        __syncthreads();
    }
    if (t < no && sel[t]) {
        int rank = 0;
        for (int j = 0; j < t; ++j) rank += sel[j];
        idxmap[(size_t)b * k + rank] = t;
    }
}

// ---------------- gather pruned tokens (cls kept at slot 0)
__global__ void gather_tokens(const float* __restrict__ hin, const float* __restrict__ rin,
                              int Lold, const int* __restrict__ idxmap, int k,
                              float* __restrict__ hout, float* __restrict__ rout,
                              int Lnew, int total)
{
    int idx = blockIdx.x * blockDim.x + threadIdx.x;
    if (idx >= total) return;
    int e = idx % 192;
    int n = idx / 192;
    int j = n % Lnew, b = n / Lnew;
    int src = (j == 0) ? 0 : 1 + idxmap[(size_t)b * k + (j - 1)];
    size_t so = ((size_t)b * Lold + src) * 192 + e;
    hout[idx] = hin[so];
    rout[idx] = rin[so];
}

__global__ void fill_ones(float* __restrict__ p, int n)
{
    int i = blockIdx.x * blockDim.x + threadIdx.x;
    if (i < n) p[i] = 1.f;
}

// ---------------------------------------------------------------------------
static inline dim3 g1(int total) { return dim3((total + 255) / 256); }

// stage A and W to padded f16, then run the WMMA GEMM
static inline void gemm(hipStream_t s, _Float16* Ah, _Float16* Wh,
                        const float* A, int lda, int M, int K,
                        const float* W, int ldw, int N,
                        const float* bias, float* C, int ldc, int epi)
{
    int Kp = (K + 31) & ~31;
    int Np = (N + 15) & ~15;
    cvt_pad_f16<<<g1(M * Kp), 256, 0, s>>>(A, lda, M, K, Ah, Kp, M * Kp);
    cvt_pad_f16<<<g1(Np * Kp), 256, 0, s>>>(W, ldw, N, K, Wh, Kp, Np * Kp);
    dim3 blk(32, 4), grd((M / 16 + 3) / 4, Np / 16);
    gemm_wmma_f16<<<grd, blk, 0, s>>>(Ah, Wh, bias, C, ldc, M, N, Kp, epi);
}

extern "C" void kernel_launch(void* const* d_in, const int* in_sizes, int n_in,
                              void* d_out, int out_size, void* d_ws, size_t ws_size,
                              hipStream_t stream)
{
    (void)in_sizes; (void)n_in; (void)out_size;
    const float* x          = (const float*)d_in[0];
    const float* patch_w    = (const float*)d_in[1];
    const float* patch_b    = (const float*)d_in[2];
    const float* cls_token  = (const float*)d_in[3];
    const float* pos_embed  = (const float*)d_in[4];
    const float* head_w     = (const float*)d_in[5];
    const float* head_b     = (const float*)d_in[6];
    const float* norm_f_w   = (const float*)d_in[7];
    const float* ln_w       = (const float*)d_in[8];
    const float* in_proj_w  = (const float*)d_in[9];
    const float* conv_w     = (const float*)d_in[10];
    const float* conv_bias  = (const float*)d_in[11];
    const float* xproj_w    = (const float*)d_in[12];
    const float* dtproj_w   = (const float*)d_in[13];
    const float* dtproj_b   = (const float*)d_in[14];
    const float* A_log      = (const float*)d_in[15];
    const float* Dv         = (const float*)d_in[16];
    const float* convr_w    = (const float*)d_in[17];
    const float* convr_bias = (const float*)d_in[18];
    const float* xprojr_w   = (const float*)d_in[19];
    const float* dtprojr_w  = (const float*)d_in[20];
    const float* dtprojr_b  = (const float*)d_in[21];
    const float* A_r_log    = (const float*)d_in[22];
    const float* D_r        = (const float*)d_in[23];
    const float* out_proj_w = (const float*)d_in[24];
    const float* pred_ln_w  = (const float*)d_in[25];
    const float* pred_ln_b  = (const float*)d_in[26];
    const float* pred_in_w  = (const float*)d_in[27];
    const float* pred_in_b  = (const float*)d_in[28];
    const float* pred_w1    = (const float*)d_in[29];
    const float* pred_b1    = (const float*)d_in[30];
    const float* pred_w2    = (const float*)d_in[31];
    const float* pred_b2    = (const float*)d_in[32];
    const float* pred_w3    = (const float*)d_in[33];
    const float* pred_b3    = (const float*)d_in[34];
    float* out = (float*)d_out;

    // ---- workspace carve-up ----------------------------------------------
    const size_t HID = (size_t)128 * 197 * 192;   // 4,841,472 floats
    const size_t XZs = (size_t)128 * 197 * 768;
    const size_t XSs = (size_t)128 * 197 * 384;
    const size_t DBs = (size_t)128 * 197 * 44;
    const size_t AHs = (size_t)25088 * 768;       // halves (biggest staged A)
    const size_t WHs = (size_t)1008 * 192;        // halves (biggest staged W = head)

    float* w0 = (float*)d_ws;
    float* hidA = w0;  w0 += HID;
    float* hidB = w0;  w0 += HID;
    float* resA = w0;  w0 += HID;
    float* resB = w0;  w0 += HID;
    float* hnorm= w0;  w0 += HID;
    float* xz   = w0;  w0 += XZs;
    float* xs   = w0;  w0 += XSs;
    float* dbc  = w0;  w0 += DBs;
    float* dt   = w0;  w0 += XSs;
    float* yf   = w0;  w0 += XSs;
    float* yb   = w0;  w0 += XSs;
    float* yy   = w0;  w0 += XSs;
    float* h1   = w0;  w0 += (size_t)25088 * 96;
    float* h2   = w0;  w0 += (size_t)25088 * 48;
    float* lg2  = w0;  w0 += (size_t)25088 * 2;
    float* scores = w0; w0 += (size_t)128 * 196;
    float* glob = w0;  w0 += (size_t)128 * 96;
    int*  idxmap = (int*)w0; w0 += (size_t)128 * 137;
    _Float16* Ah = (_Float16*)w0; w0 += (AHs + 1) / 2;
    _Float16* Wh = (_Float16*)w0; w0 += (WHs + 1) / 2;
    if ((size_t)((char*)w0 - (char*)d_ws) > ws_size) return;  // ws too small: bail

    // aliases only live while xz is dead (predictor stage)
    float* tok  = xs;
    float* xp0  = xz;
    float* xp   = xz + (size_t)5000000;
    float* feat = xz + (size_t)10000000;

    // ---- patch embedding: im2col straight to f16 staging ------------------
    im2col_patch_f16<<<g1(25088 * 768), 256, 0, stream>>>(x, Ah, 25088 * 768);
    cvt_pad_f16<<<g1(192 * 768), 256, 0, stream>>>(patch_w, 768, 192, 768, Wh, 768, 192 * 768);
    {
        dim3 blk(32, 4), grd((25088 / 16 + 3) / 4, 192 / 16);
        gemm_wmma_f16<<<grd, blk, 0, stream>>>(Ah, Wh, patch_b, tok, 192,
                                               25088, 192, 768, EPI_BIAS);
    }
    assemble_hidden<<<g1(128 * 197 * 192), 256, 0, stream>>>(tok, cls_token, pos_embed,
                                                             hidA, 128 * 197 * 192);

    float* hid = hidA; float* hAlt = hidB;
    float* res = resA; float* rAlt = resB;
    int L = 197, pc = 0;

    for (int i = 0; i < 24; ++i) {
        if (i == 6 || i == 12 || i == 18) {
            int no = L - 1;
            int k  = (i == 6) ? 137 : (i == 12) ? 96 : 67;
            int Bo = 128 * no;
            pred_layernorm<<<dim3((Bo + 7) / 8), 256, 0, stream>>>(
                hid, res, L, pred_ln_w + pc * 192, pred_ln_b + pc * 192, xp0, Bo);
            gemm(stream, Ah, Wh, xp0, 192, Bo, 192,
                 pred_in_w + (size_t)pc * 192 * 192, 192, 192,
                 pred_in_b + pc * 192, xp, 192, EPI_GELU);
            pred_pool<<<g1(128 * 96), 256, 0, stream>>>(xp, no, glob, 128 * 96);
            pred_feat<<<g1(Bo * 192), 256, 0, stream>>>(xp, glob, no, feat, Bo * 192);
            gemm(stream, Ah, Wh, feat, 192, Bo, 192,
                 pred_w1 + (size_t)pc * 96 * 192, 192, 96,
                 pred_b1 + pc * 96, h1, 96, EPI_GELU);
            gemm(stream, Ah, Wh, h1, 96, Bo, 96,
                 pred_w2 + (size_t)pc * 48 * 96, 96, 48,
                 pred_b2 + pc * 48, h2, 48, EPI_GELU);
            gemm(stream, Ah, Wh, h2, 48, Bo, 48,
                 pred_w3 + (size_t)pc * 2 * 48, 48, 2,
                 pred_b3 + pc * 2, lg2, 2, EPI_BIAS);
            pred_score<<<g1(Bo), 256, 0, stream>>>(lg2, scores, Bo);
            topk_sorted<<<dim3(128), 256, 0, stream>>>(scores, no, k, idxmap);
            int Lnew = k + 1;
            gather_tokens<<<g1(128 * Lnew * 192), 256, 0, stream>>>(
                hid, res, L, idxmap, k, hAlt, rAlt, Lnew, 128 * Lnew * 192);
            { float* t_ = hid; hid = hAlt; hAlt = t_; }
            { float* t_ = res; res = rAlt; rAlt = t_; }
            L = Lnew; ++pc;
        }

        const int N = 128 * L;
        add_rmsnorm<<<dim3((N + 7) / 8), 256, 0, stream>>>(hid, res, ln_w + i * 192,
                                                           hnorm, N, i == 0);
        gemm(stream, Ah, Wh, hnorm, 192, N, 192,
             in_proj_w + (size_t)i * 768 * 192, 192, 768,
             nullptr, xz, 768, EPI_NONE);

        // forward branch
        conv_silu<<<g1(N * 384), 256, 0, stream>>>(xz, L, conv_w + (size_t)i * 1536,
                                                   conv_bias + i * 384, xs, N * 384, 0);
        gemm(stream, Ah, Wh, xs, 384, N, 384,
             xproj_w + (size_t)i * 44 * 384, 384, 44, nullptr, dbc, 44, EPI_NONE);
        gemm(stream, Ah, Wh, dbc, 44, N, 12,
             dtproj_w + (size_t)i * 384 * 12, 12, 384,
             dtproj_b + i * 384, dt, 384, EPI_SOFTPLUS);
        ssm_scan<<<g1(128 * 384), 256, 0, stream>>>(xs, dbc, dt, A_log + (size_t)i * 6144,
                                                    Dv + i * 384, yf, L);
        // backward branch (sequence reversed)
        conv_silu<<<g1(N * 384), 256, 0, stream>>>(xz, L, convr_w + (size_t)i * 1536,
                                                   convr_bias + i * 384, xs, N * 384, 1);
        gemm(stream, Ah, Wh, xs, 384, N, 384,
             xprojr_w + (size_t)i * 44 * 384, 384, 44, nullptr, dbc, 44, EPI_NONE);
        gemm(stream, Ah, Wh, dbc, 44, N, 12,
             dtprojr_w + (size_t)i * 384 * 12, 12, 384,
             dtprojr_b + i * 384, dt, 384, EPI_SOFTPLUS);
        ssm_scan<<<g1(128 * 384), 256, 0, stream>>>(xs, dbc, dt, A_r_log + (size_t)i * 6144,
                                                    D_r + i * 384, yb, L);

        combine_gate<<<g1(N * 384), 256, 0, stream>>>(yf, yb, xz, L, yy, N * 384);
        gemm(stream, Ah, Wh, yy, 384, N, 384,
             out_proj_w + (size_t)i * 192 * 384, 384, 192, nullptr, hid, 192, EPI_NONE);
    }

    // ---- final norm + classifier head (cls rows selected via lda stride) --
    add_rmsnorm<<<dim3((128 * L + 7) / 8), 256, 0, stream>>>(hid, res, norm_f_w,
                                                             hnorm, 128 * L, 0);
    gemm(stream, Ah, Wh, hnorm, L * 192, 128, 192,
         head_w, 192, 1000, head_b, out, 1000, EPI_BIAS);
    fill_ones<<<g1(24 * 128), 256, 0, stream>>>(out + 128 * 1000, 24 * 128);
}
